// SpikingDQNv1_18743237279794
// MI455X (gfx1250) — matrix operand order repaired and checked by reading, not compiled, not run on hardware
//
#include <hip/hip_runtime.h>
#include <hip/hip_bf16.h>

typedef __attribute__((ext_vector_type(8)))  float  v8f;
typedef __attribute__((ext_vector_type(8)))  __bf16 v8bf;
typedef __attribute__((ext_vector_type(16))) __bf16 v16bf;

// GCC-style vector to exactly match the async-LDS builtin's parameter type:
//   param0: 'int __attribute__((vector_size(16))) __device__ *'  (addrspace 1)
//   param1: LDS side (addrspace 3)
typedef int async_v4i __attribute__((vector_size(16)));
typedef __attribute__((address_space(1))) async_v4i* gas_v4i_p;
typedef __attribute__((address_space(3))) async_v4i* las_v4i_p;

#define B_ROWS 8192
#define IN_FEA 4132
#define HIDDEN 1024
#define OUT_N  9
#define NSTEPS 10
#define BETA_F 0.9f
#define THR_F  1.0f

// Probe-safe gate for the gfx1250 async global->LDS path (ASYNCcnt tracked).
#if defined(__gfx1250__) &&                                             \
    __has_builtin(__builtin_amdgcn_global_load_async_to_lds_b128) &&    \
    __has_builtin(__builtin_amdgcn_s_wait_asynccnt)
#define USE_ASYNC_LDS 1
#else
#define USE_ASYNC_LDS 0
#endif

__device__ __forceinline__ v16bf cat8(v8bf a, v8bf b) {
  return __builtin_shufflevector(a, b, 0, 1, 2, 3, 4, 5, 6, 7, 8, 9, 10, 11,
                                 12, 13, 14, 15);
}

__device__ __forceinline__ v8f wmma_bf16(v16bf a, v16bf b, v8f c) {
  return __builtin_amdgcn_wmma_f32_16x16x32_bf16(false, a, false, b, (short)0,
                                                 c, false, false);
}

// ---------------------------------------------------------------------------
// W2 (fp32) -> hi/lo bf16 planes: W2 ~= hi + lo to ~2^-16 relative accuracy.
// spk1 is exactly {0,1} in bf16, so spk1 @ (hi+lo) == spk1 @ W2 to fp32-ish.
// ---------------------------------------------------------------------------
__global__ __launch_bounds__(256) void k_convert_w2(
    const float* __restrict__ W2, __bf16* __restrict__ h, __bf16* __restrict__ l) {
  int i = blockIdx.x * 256 + threadIdx.x;
  if (i >= HIDDEN * HIDDEN) return;
  float w = W2[i];
  __bf16 hi = (__bf16)w;
  h[i] = hi;
  l[i] = (__bf16)(w - (float)hi);
}

__global__ __launch_bounds__(256) void k_zero(float* __restrict__ p, int n) {
  int i = blockIdx.x * 256 + threadIdx.x;
  if (i < n) p[i] = 0.0f;
}

// ---------------------------------------------------------------------------
// GEMM1: l1[B,HID] = x[B,INF] @ W1[HID,INF]^T + b1 via bf16x3 split on the
// bf16 matrix pipe (K=32/instr vs K=4 for the f32 WMMA -> ~2.7x fewer matrix
// issue slots). hi/lo conversion done once per element at LDS-fill time
// (VALU co-executes with XDL WMMA).
// Block: 256 thr = 8 waves; tile 128M x 64N; wave = 16M x 64N (4 accums).
// bf16 A 16x32, k order used here: lanes0-15 row M, K=0..15; lanes16-31 K=16..31.
// bf16 B 32x16, same k order: lanes0-15 col N K=0..15; lanes16-31 K=16..31.
// ---------------------------------------------------------------------------
__global__ __launch_bounds__(256) void k_gemm1(
    const float* __restrict__ x, const float* __restrict__ W1,
    const float* __restrict__ b1, float* __restrict__ l1) {
  __shared__ __align__(64) __bf16 sXh[128][32];
  __shared__ __align__(64) __bf16 sXl[128][32];
  __shared__ __align__(64) __bf16 sWh[64][32];
  __shared__ __align__(64) __bf16 sWl[64][32];
  const int tid  = threadIdx.x;
  const int wave = tid >> 5, lane = tid & 31;
  const int l15  = lane & 15, hsel = lane >> 4;
  const int bm = blockIdx.x * 128;
  const int bn = blockIdx.y * 64;
  v8f acc[4] = {};
  const int nKB = (IN_FEA + 31) / 32;  // 130, last slab zero-padded
  for (int kb = 0; kb < nKB; ++kb) {
    const int k0 = kb * 32;
#pragma unroll
    for (int i = 0; i < 16; ++i) {          // 128x32 x-tile, split hi/lo
      int idx = tid + i * 256;
      int r = idx >> 5, c = idx & 31, gk = k0 + c;
      float v = (gk < IN_FEA) ? x[(size_t)(bm + r) * IN_FEA + gk] : 0.0f;
      __bf16 hi = (__bf16)v;
      sXh[r][c] = hi;
      sXl[r][c] = (__bf16)(v - (float)hi);
    }
#pragma unroll
    for (int i = 0; i < 8; ++i) {           // 64x32 w1-tile, split hi/lo
      int idx = tid + i * 256;
      int r = idx >> 5, c = idx & 31, gk = k0 + c;
      float v = (gk < IN_FEA) ? W1[(size_t)(bn + r) * IN_FEA + gk] : 0.0f;
      __bf16 hi = (__bf16)v;
      sWh[r][c] = hi;
      sWl[r][c] = (__bf16)(v - (float)hi);
    }
    __syncthreads();
    if (kb + 1 < nKB) {                     // warm next slab (global_prefetch_b8)
      __builtin_prefetch(&x[(size_t)(bm + (tid >> 1)) * IN_FEA + k0 + 32], 0, 0);
    }
    const int arow = wave * 16 + l15;
    v16bf Ah = cat8(*(const v8bf*)&sXh[arow][hsel * 16],
                    *(const v8bf*)&sXh[arow][hsel * 16 + 8]);
    v16bf Al = cat8(*(const v8bf*)&sXl[arow][hsel * 16],
                    *(const v8bf*)&sXl[arow][hsel * 16 + 8]);
#pragma unroll
    for (int j = 0; j < 4; ++j) {
      const int bcol = j * 16 + l15;
      v16bf Bh = *(const v16bf*)&sWh[bcol][hsel * 16];
      v16bf Bl = *(const v16bf*)&sWl[bcol][hsel * 16];
      acc[j] = wmma_bf16(Ah, Bh, acc[j]);   // hi*hi
      acc[j] = wmma_bf16(Ah, Bl, acc[j]);   // hi*lo
      acc[j] = wmma_bf16(Al, Bh, acc[j]);   // lo*hi  (lo*lo ~ 2^-32, dropped)
    }
    __syncthreads();
  }
  // C/D layout: VGPR r -> M = r (lanes 0-15) / r+8 (lanes 16-31), N = lane&15
#pragma unroll
  for (int j = 0; j < 4; ++j) {
    const int col  = bn + j * 16 + l15;
    const float bb = b1[col];
#pragma unroll
    for (int r = 0; r < 8; ++r) {
      const int row = bm + wave * 16 + r + hsel * 8;
      l1[(size_t)row * HIDDEN + col] = acc[j][r] + bb;
    }
  }
}

// ---------------------------------------------------------------------------
// LIF layer 1 (elementwise): subtract-reset from previous mem, heaviside spike
// emitted as exact bf16 {0,1} for the tensor-core recurrent GEMM.
// ---------------------------------------------------------------------------
__global__ __launch_bounds__(256) void k_lif1(
    const float* __restrict__ l1, float* __restrict__ mem1,
    __bf16* __restrict__ spk1) {
  size_t i = (size_t)blockIdx.x * 256 + threadIdx.x;
  if (i >= (size_t)B_ROWS * HIDDEN) return;
  float m   = mem1[i];
  float rst = (m > THR_F) ? THR_F : 0.0f;
  m = BETA_F * m + l1[i] - rst;
  mem1[i] = m;
  spk1[i] = (__bf16)(((m - THR_F) > 0.0f) ? 1.0f : 0.0f);
}

// ---------------------------------------------------------------------------
// GEMM2 + fused LIF2: l2 = spk1 @ W2^T + b2, two accumulating bf16 WMMAs per
// tile-step (hi + lo planes of W2). K-slab = 64 (32KB LDS) to halve barriers.
// LDS fills use GLOBAL_LOAD_ASYNC_TO_LDS_B128 + s_wait_asynccnt when the
// toolchain exposes them; 16B vector copies otherwise.
// ---------------------------------------------------------------------------
__global__ __launch_bounds__(256) void k_gemm2_lif2(
    const __bf16* __restrict__ spk1, const __bf16* __restrict__ W2h,
    const __bf16* __restrict__ W2l, const float* __restrict__ b2,
    float* __restrict__ mem2, float* __restrict__ spk_sum) {
  __shared__ __align__(64) __bf16 sA[128][64];
  __shared__ __align__(64) __bf16 sBh[64][64];
  __shared__ __align__(64) __bf16 sBl[64][64];
  const int tid  = threadIdx.x;
  const int wave = tid >> 5, lane = tid & 31;
  const int l15  = lane & 15, hsel = lane >> 4;
  const int bm = blockIdx.x * 128;
  const int bn = blockIdx.y * 64;
  v8f acc[4] = {};
  for (int kb = 0; kb < HIDDEN / 64; ++kb) {
    const int k0 = kb * 64;
    const __bf16* gA  = spk1 + (size_t)bm * HIDDEN + k0;
    const __bf16* gBh = W2h  + (size_t)bn * HIDDEN + k0;
    const __bf16* gBl = W2l  + (size_t)bn * HIDDEN + k0;
#if USE_ASYNC_LDS
#pragma unroll
    for (int t = 0; t < 4; ++t) {           // 128x64 spike tile, 16B chunks
      int cid = tid + t * 256;
      int r = cid >> 3, c = (cid & 7) * 8;
      __builtin_amdgcn_global_load_async_to_lds_b128(
          (gas_v4i_p)(gA + (size_t)r * HIDDEN + c),
          (las_v4i_p)&sA[r][c], 0, 0);
    }
#pragma unroll
    for (int t = 0; t < 2; ++t) {           // 64x64 weight tiles (hi, lo)
      int cid = tid + t * 256;
      int r = cid >> 3, c = (cid & 7) * 8;
      __builtin_amdgcn_global_load_async_to_lds_b128(
          (gas_v4i_p)(gBh + (size_t)r * HIDDEN + c),
          (las_v4i_p)&sBh[r][c], 0, 0);
      __builtin_amdgcn_global_load_async_to_lds_b128(
          (gas_v4i_p)(gBl + (size_t)r * HIDDEN + c),
          (las_v4i_p)&sBl[r][c], 0, 0);
    }
    __builtin_amdgcn_s_wait_asynccnt(0);
    __syncthreads();
#else
#pragma unroll
    for (int t = 0; t < 4; ++t) {
      int cid = tid + t * 256;
      int r = cid >> 3, c = (cid & 7) * 8;
      *(v8bf*)&sA[r][c] = *(const v8bf*)(gA + (size_t)r * HIDDEN + c);
    }
#pragma unroll
    for (int t = 0; t < 2; ++t) {
      int cid = tid + t * 256;
      int r = cid >> 3, c = (cid & 7) * 8;
      *(v8bf*)&sBh[r][c] = *(const v8bf*)(gBh + (size_t)r * HIDDEN + c);
      *(v8bf*)&sBl[r][c] = *(const v8bf*)(gBl + (size_t)r * HIDDEN + c);
    }
    __syncthreads();
#endif
    const int arow = wave * 16 + l15;
#pragma unroll
    for (int ks = 0; ks < 2; ++ks) {        // two K=32 steps per slab
      const int kk = ks * 32;
      v16bf a = cat8(*(const v8bf*)&sA[arow][kk + hsel * 16],
                     *(const v8bf*)&sA[arow][kk + hsel * 16 + 8]);
#pragma unroll
      for (int j = 0; j < 4; ++j) {
        const int bcol = j * 16 + l15;
        v16bf bh = *(const v16bf*)&sBh[bcol][kk + hsel * 16];
        v16bf bl = *(const v16bf*)&sBl[bcol][kk + hsel * 16];
        acc[j] = wmma_bf16(a, bh, acc[j]);
        acc[j] = wmma_bf16(a, bl, acc[j]);
      }
    }
    __syncthreads();
  }
  // fused LIF2 epilogue (each output element owned by exactly one wave)
#pragma unroll
  for (int j = 0; j < 4; ++j) {
    const int col  = bn + j * 16 + l15;
    const float bb = b2[col];
#pragma unroll
    for (int r = 0; r < 8; ++r) {
      const int row = bm + wave * 16 + r + hsel * 8;
      const size_t idx = (size_t)row * HIDDEN + col;
      float m   = mem2[idx];
      float rst = (m > THR_F) ? THR_F : 0.0f;
      m = BETA_F * m + (acc[j][r] + bb) - rst;
      mem2[idx] = m;
      if ((m - THR_F) > 0.0f) spk_sum[idx] += 1.0f;
    }
  }
}

// ---------------------------------------------------------------------------
// q = spk_sum @ W3^T + b3  -> [8192, 9]; 0.15 GFLOP, plain VALU is fine.
// ---------------------------------------------------------------------------
__global__ __launch_bounds__(256) void k_gemm3(
    const float* __restrict__ ssum, const float* __restrict__ W3,
    const float* __restrict__ b3, float* __restrict__ q) {
  const int row = blockIdx.x * 256 + threadIdx.x;
  if (row >= B_ROWS) return;
  float acc[OUT_N] = {};
  const float* sr = ssum + (size_t)row * HIDDEN;
  for (int k = 0; k < HIDDEN; ++k) {
    float s = sr[k];
#pragma unroll
    for (int o = 0; o < OUT_N; ++o) acc[o] += s * W3[(size_t)o * HIDDEN + k];
  }
#pragma unroll
  for (int o = 0; o < OUT_N; ++o) q[(size_t)row * OUT_N + o] = acc[o] + b3[o];
}

// ---------------------------------------------------------------------------
extern "C" void kernel_launch(void* const* d_in, const int* in_sizes, int n_in,
                              void* d_out, int out_size, void* d_ws,
                              size_t ws_size, hipStream_t stream) {
  const float* x  = (const float*)d_in[0];
  const float* W1 = (const float*)d_in[1];
  const float* b1 = (const float*)d_in[2];
  const float* W2 = (const float*)d_in[3];
  const float* b2 = (const float*)d_in[4];
  const float* W3 = (const float*)d_in[5];
  const float* b3 = (const float*)d_in[6];

  const size_t NBH = (size_t)B_ROWS * HIDDEN;  // 8,388,608
  char* ws = (char*)d_ws;
  size_t off = 0;
  float* l1      = (float*)(ws + off); off += NBH * 4;
  float* mem1    = (float*)(ws + off); off += NBH * 4;   // | contiguous zero
  float* mem2    = (float*)(ws + off); off += NBH * 4;   // | region (3*NBH)
  float* spk_sum = (float*)(ws + off); off += NBH * 4;   // |
  __bf16* spk1   = (__bf16*)(ws + off); off += NBH * 2;
  __bf16* W2h    = (__bf16*)(ws + off); off += (size_t)HIDDEN * HIDDEN * 2;
  __bf16* W2l    = (__bf16*)(ws + off); off += (size_t)HIDDEN * HIDDEN * 2;
  (void)ws_size; (void)in_sizes; (void)n_in; (void)out_size;

  const dim3 blk(256);
  const dim3 gGemm(B_ROWS / 128, HIDDEN / 64);            // 64 x 16

  k_convert_w2<<<(HIDDEN * HIDDEN + 255) / 256, blk, 0, stream>>>(W2, W2h, W2l);
  k_zero<<<(int)((3 * NBH + 255) / 256), blk, 0, stream>>>(mem1, (int)(3 * NBH));
  k_gemm1<<<gGemm, blk, 0, stream>>>(x, W1, b1, l1);
  for (int t = 0; t < NSTEPS; ++t) {
    k_lif1<<<(int)((NBH + 255) / 256), blk, 0, stream>>>(l1, mem1, spk1);
    k_gemm2_lif2<<<gGemm, blk, 0, stream>>>(spk1, W2h, W2l, b2, mem2, spk_sum);
  }
  k_gemm3<<<(B_ROWS + 255) / 256, blk, 0, stream>>>(spk_sum, W3, b3,
                                                    (float*)d_out);
}
